// TransformerBlock_89507118449220
// MI455X (gfx1250) — matrix operand order repaired and checked by reading, not compiled, not run on hardware
//
#include <hip/hip_runtime.h>
#include <stdint.h>

#define HID 1024
#define NHEAD 8
#define HDIM 128
#define SEQ 2048
#define BATCH 2
#define TOK (BATCH * SEQ)   // 4096
#define FF (4 * HID)        // 4096

typedef __attribute__((ext_vector_type(16))) __bf16 v16bf;
typedef __attribute__((ext_vector_type(8)))  float  v8f;

union AB32 {                 // one WMMA 16-bit operand (16 halves / lane = 8 VGPRs)
    uint4 q[2];
    unsigned short h[16];
    v16bf v;
};

__device__ __forceinline__ unsigned short f2bf(float f) {
    unsigned int u = __float_as_uint(f);
    unsigned int r = u + 0x7FFFu + ((u >> 16) & 1u);   // round-to-nearest-even
    if ((u & 0x7F800000u) == 0x7F800000u) r = u;        // inf/nan: truncate
    return (unsigned short)(r >> 16);
}

__device__ __forceinline__ float redmax16(float x) {
    x = fmaxf(x, __shfl_xor(x, 1, 32));
    x = fmaxf(x, __shfl_xor(x, 2, 32));
    x = fmaxf(x, __shfl_xor(x, 4, 32));
    x = fmaxf(x, __shfl_xor(x, 8, 32));
    return x;
}
__device__ __forceinline__ float redsum16(float x) {
    x += __shfl_xor(x, 1, 32);
    x += __shfl_xor(x, 2, 32);
    x += __shfl_xor(x, 4, 32);
    x += __shfl_xor(x, 8, 32);
    return x;
}

// ---------------- weight fp32 [K,N] -> bf16 transposed [N,K] ----------------
__global__ void wt_kernel(const float* __restrict__ W, unsigned short* __restrict__ Wt,
                          int K, int N) {
    long long idx = (long long)blockIdx.x * blockDim.x + threadIdx.x;
    if (idx >= (long long)K * N) return;
    int n = (int)(idx / K);
    int k = (int)(idx % K);
    Wt[idx] = f2bf(W[(long long)k * N + n]);
}

// ---------------- LayerNorm: 1 wave per token row, fp32 in -> bf16 out ------
__global__ void ln_kernel(const float* __restrict__ x, const float* __restrict__ g,
                          const float* __restrict__ b, unsigned short* __restrict__ out) {
    const int row  = blockIdx.x;
    const int lane = threadIdx.x;
    const float* xr = x + (long long)row * HID;
    float v[32];
    float s = 0.f;
#pragma unroll
    for (int i = 0; i < 32; ++i) { v[i] = xr[lane * 32 + i]; s += v[i]; }
#pragma unroll
    for (int m = 1; m < 32; m <<= 1) s += __shfl_xor(s, m, 32);
    const float mu = s * (1.0f / HID);
    float q = 0.f;
#pragma unroll
    for (int i = 0; i < 32; ++i) { float d = v[i] - mu; q += d * d; }
#pragma unroll
    for (int m = 1; m < 32; m <<= 1) q += __shfl_xor(q, m, 32);
    const float inv = rsqrtf(q * (1.0f / HID) + 1e-5f);
    unsigned short* orow = out + (long long)row * HID + lane * 32;
#pragma unroll
    for (int i = 0; i < 32; ++i)
        orow[i] = f2bf((v[i] - mu) * inv * g[lane * 32 + i] + b[lane * 32 + i]);
}

// ---------------- WMMA bf16 GEMM: C = act(A[M,K] @ Bt[N,K]^T + bias) --------
// mode 0: bf16 out; mode 1: GELU(exact) bf16 out; mode 2: f32 out = res + val
__global__ void gemm_kernel(const unsigned short* __restrict__ A,
                            const unsigned short* __restrict__ Bt,
                            const float* __restrict__ bias,
                            const float* __restrict__ res,
                            float* __restrict__ outf,
                            unsigned short* __restrict__ outh,
                            int M, int N, int K, int mode) {
    const int lane = threadIdx.x & 31;
    const int wave = threadIdx.x >> 5;
    const int half = lane >> 4;     // 0: lanes 0-15, 1: lanes 16-31
    const int l16  = lane & 15;
    const int m0 = blockIdx.y * 128 + wave * 16;
    const int n0 = blockIdx.x * 64;
    (void)M;

    v8f acc[4];
#pragma unroll
    for (int j = 0; j < 4; ++j) acc[j] = (v8f){0.f, 0.f, 0.f, 0.f, 0.f, 0.f, 0.f, 0.f};

    // A layout: lane = M row (l16); halves K = {0..7,16..23} (half=0) / {8..15,24..31}
    const unsigned short* arow = A + (long long)(m0 + l16) * K + half * 8;
    for (int kb = 0; kb < K; kb += 32) {
        AB32 a;
        a.q[0] = *(const uint4*)(arow + kb);
        a.q[1] = *(const uint4*)(arow + kb + 16);
#pragma unroll
        for (int j = 0; j < 4; ++j) {
            // B layout: lane = N col (l16); halves K = {0..7,16..23} (half=0) / {8..15,24..31}
            const unsigned short* brow =
                Bt + (long long)(n0 + j * 16 + l16) * K + kb + half * 8;
            AB32 bm;
            bm.q[0] = *(const uint4*)(brow);
            bm.q[1] = *(const uint4*)(brow + 16);
            acc[j] = __builtin_amdgcn_wmma_f32_16x16x32_bf16(
                false, a.v, false, bm.v, (short)0, acc[j], false, false);
        }
    }

#pragma unroll
    for (int j = 0; j < 4; ++j) {
        const int n = n0 + j * 16 + l16;
        const float bb = bias[n];
#pragma unroll
        for (int e = 0; e < 8; ++e) {
            const int m = m0 + e + half * 8;    // C layout: VGPR e -> row e (+8 upper half)
            float val = acc[j][e] + bb;
            if (mode == 1) val = 0.5f * val * (1.0f + erff(val * 0.70710678118654752f));
            const long long o = (long long)m * N + n;
            if (mode == 2) outf[o] = res[o] + val;
            else           outh[o] = f2bf(val);
        }
    }
}

// ---------------- Flash attention: 1 wave per (b, head, 16-query tile) -----
__global__ void attn_kernel(const unsigned short* __restrict__ qm,
                            const unsigned short* __restrict__ km,
                            const unsigned short* __restrict__ vm,
                            unsigned short* __restrict__ ym) {
    __shared__ __align__(16) unsigned short pbuf[16 * 32];   // P tile C->A relayout
    const int lane = threadIdx.x & 31;
    const int half = lane >> 4;
    const int l16  = lane & 15;
    const int qt   = blockIdx.x;
    const int hidx = blockIdx.y;
    const int b    = blockIdx.z;
    const int qs   = qt * 16;
    const long long rowbase = (long long)b * SEQ;
    const int hoff = hidx * HDIM;
    const float scale = 0.08838834764831845f;   // 1/sqrt(128)

    // Q tile in A layout, all 128 head dims (4 K-chunks of 32)
    AB32 qa[4];
    {
        const unsigned short* qr = qm + (rowbase + qs + l16) * HID + hoff + half * 8;
#pragma unroll
        for (int dc = 0; dc < 4; ++dc) {
            qa[dc].q[0] = *(const uint4*)(qr + dc * 32);
            qa[dc].q[1] = *(const uint4*)(qr + dc * 32 + 16);
        }
    }

    v8f yacc[8];
#pragma unroll
    for (int dn = 0; dn < 8; ++dn) yacc[dn] = (v8f){0.f,0.f,0.f,0.f,0.f,0.f,0.f,0.f};
    float rowM[8], rowL[8];
#pragma unroll
    for (int e = 0; e < 8; ++e) { rowM[e] = -1e30f; rowL[e] = 0.f; }

    const int nch = (qs + 47) >> 5;        // causal: only chunks with key <= qs+15
    for (int c = 0; c < nch; ++c) {
        const int jb = c * 32;
        // ---- scores: S = Q @ K^T for 2 key subtiles of 16 ----
        v8f s[2];
#pragma unroll
        for (int sub = 0; sub < 2; ++sub) {
            const int ks = jb + sub * 16 + l16;
            const int kr = (ks < SEQ) ? ks : (SEQ - 1);
            const unsigned short* krow = km + (rowbase + kr) * HID + hoff + half * 8;
            v8f sc = (v8f){0.f,0.f,0.f,0.f,0.f,0.f,0.f,0.f};
#pragma unroll
            for (int dc = 0; dc < 4; ++dc) {
                AB32 bm;
                bm.q[0] = *(const uint4*)(krow + dc * 32);
                bm.q[1] = *(const uint4*)(krow + dc * 32 + 16);
                sc = __builtin_amdgcn_wmma_f32_16x16x32_bf16(
                    false, qa[dc].v, false, bm.v, (short)0, sc, false, false);
            }
            s[sub] = sc;
        }
        // ---- causal mask + flash running softmax ----
        float s0v[8], s1v[8], mx[8], sclv[8];
#pragma unroll
        for (int e = 0; e < 8; ++e) {
            const int qrow = qs + e + half * 8;
            float a0 = s[0][e] * scale;
            float a1 = s[1][e] * scale;
            if (jb + l16      > qrow) a0 = -1e30f;
            if (jb + 16 + l16 > qrow) a1 = -1e30f;
            s0v[e] = a0; s1v[e] = a1;
            mx[e] = redmax16(fmaxf(a0, a1));
        }
#pragma unroll
        for (int e = 0; e < 8; ++e) {
            const float nm  = fmaxf(rowM[e], mx[e]);
            const float scl = __expf(rowM[e] - nm);
            const float p0  = __expf(s0v[e] - nm);
            const float p1  = __expf(s1v[e] - nm);
            rowL[e] = rowL[e] * scl + redsum16(p0 + p1);
            rowM[e] = nm;
            sclv[e] = scl;
            pbuf[(e + half * 8) * 32 + l16]      = f2bf(p0);
            pbuf[(e + half * 8) * 32 + 16 + l16] = f2bf(p1);
        }
#pragma unroll
        for (int dn = 0; dn < 8; ++dn)
#pragma unroll
            for (int e = 0; e < 8; ++e) yacc[dn][e] *= sclv[e];

        asm volatile("s_wait_dscnt 0" ::: "memory");   // LDS stores visible in-wave
        // P tile back out in A layout: lane = query row, K halves per spec
        AB32 pa;
        pa.q[0] = *(const uint4*)&pbuf[l16 * 32 + half * 8];
        pa.q[1] = *(const uint4*)&pbuf[l16 * 32 + 16 + half * 8];
        asm volatile("s_wait_dscnt 0" ::: "memory");

        // ---- Y += P @ V ; V gathered as B operand (lane = d col, 16 keys) ----
#pragma unroll
        for (int dn = 0; dn < 8; ++dn) {
            AB32 bv;
#pragma unroll
            for (int i = 0; i < 16; ++i) {
                const int ks = jb + half * 8 + (i & 7) + (i >> 3) * 16;
                const int kr = (ks < SEQ) ? ks : (SEQ - 1);
                bv.h[i] = vm[(rowbase + kr) * HID + hoff + dn * 16 + l16];
            }
            yacc[dn] = __builtin_amdgcn_wmma_f32_16x16x32_bf16(
                false, pa.v, false, bv.v, (short)0, yacc[dn], false, false);
        }
    }

    // ---- write Y tile (merge heads by column offset), bf16 for Wo GEMM ----
#pragma unroll
    for (int dn = 0; dn < 8; ++dn)
#pragma unroll
        for (int e = 0; e < 8; ++e) {
            const long long r = rowbase + qs + e + half * 8;
            ym[r * HID + hoff + dn * 16 + l16] = f2bf(yacc[dn][e] / rowL[e]);
        }
}

// ---------------------------------------------------------------------------
extern "C" void kernel_launch(void* const* d_in, const int* in_sizes, int n_in,
                              void* d_out, int out_size, void* d_ws, size_t ws_size,
                              hipStream_t stream) {
    (void)in_sizes; (void)n_in; (void)out_size; (void)ws_size;
    const float* x    = (const float*)d_in[0];
    const float* ln1g = (const float*)d_in[1];
    const float* ln1b = (const float*)d_in[2];
    const float* ln2g = (const float*)d_in[3];
    const float* ln2b = (const float*)d_in[4];
    const float* Wq = (const float*)d_in[5];   const float* bq = (const float*)d_in[6];
    const float* Wk = (const float*)d_in[7];   const float* bk = (const float*)d_in[8];
    const float* Wv = (const float*)d_in[9];   const float* bv = (const float*)d_in[10];
    const float* Wo = (const float*)d_in[11];  const float* bo = (const float*)d_in[12];
    const float* W1 = (const float*)d_in[13];  const float* b1 = (const float*)d_in[14];
    const float* W2 = (const float*)d_in[15];  const float* b2 = (const float*)d_in[16];

    char* ws = (char*)d_ws;
    size_t off = 0;
    auto alloc = [&](size_t bytes) -> char* {
        char* p = ws + off;
        off += (bytes + 255) & ~(size_t)255;
        return p;
    };
    unsigned short* Wq_t = (unsigned short*)alloc((size_t)HID * HID * 2);
    unsigned short* Wk_t = (unsigned short*)alloc((size_t)HID * HID * 2);
    unsigned short* Wv_t = (unsigned short*)alloc((size_t)HID * HID * 2);
    unsigned short* Wo_t = (unsigned short*)alloc((size_t)HID * HID * 2);
    unsigned short* W1_t = (unsigned short*)alloc((size_t)HID * FF * 2);
    unsigned short* W2_t = (unsigned short*)alloc((size_t)FF * HID * 2);
    unsigned short* hbuf = (unsigned short*)alloc((size_t)TOK * HID * 2);
    unsigned short* qbuf = (unsigned short*)alloc((size_t)TOK * HID * 2);
    unsigned short* kbuf = (unsigned short*)alloc((size_t)TOK * HID * 2);
    unsigned short* vbuf = (unsigned short*)alloc((size_t)TOK * HID * 2);
    unsigned short* ybuf = (unsigned short*)alloc((size_t)TOK * HID * 2);
    float*          x1   = (float*)         alloc((size_t)TOK * HID * 4);
    unsigned short* h2   = (unsigned short*)alloc((size_t)TOK * HID * 2);
    unsigned short* gbuf = (unsigned short*)alloc((size_t)TOK * FF * 2);

    // 1) weights -> bf16, transposed [N,K]
    const int sq = (HID * HID + 255) / 256;
    const int bg = (HID * FF + 255) / 256;
    wt_kernel<<<sq, 256, 0, stream>>>(Wq, Wq_t, HID, HID);
    wt_kernel<<<sq, 256, 0, stream>>>(Wk, Wk_t, HID, HID);
    wt_kernel<<<sq, 256, 0, stream>>>(Wv, Wv_t, HID, HID);
    wt_kernel<<<sq, 256, 0, stream>>>(Wo, Wo_t, HID, HID);
    wt_kernel<<<bg, 256, 0, stream>>>(W1, W1_t, HID, FF);   // K=HID, N=FF
    wt_kernel<<<bg, 256, 0, stream>>>(W2, W2_t, FF, HID);   // K=FF,  N=HID

    // 2) LN1
    ln_kernel<<<TOK, 32, 0, stream>>>(x, ln1g, ln1b, hbuf);

    // 3) Q/K/V projections
    dim3 gH(HID / 64, TOK / 128);
    gemm_kernel<<<gH, 256, 0, stream>>>(hbuf, Wq_t, bq, nullptr, nullptr, qbuf, TOK, HID, HID, 0);
    gemm_kernel<<<gH, 256, 0, stream>>>(hbuf, Wk_t, bk, nullptr, nullptr, kbuf, TOK, HID, HID, 0);
    gemm_kernel<<<gH, 256, 0, stream>>>(hbuf, Wv_t, bv, nullptr, nullptr, vbuf, TOK, HID, HID, 0);

    // 4) flash attention
    attn_kernel<<<dim3(SEQ / 16, NHEAD, BATCH), 32, 0, stream>>>(qbuf, kbuf, vbuf, ybuf);

    // 5) output projection + residual (fp32)
    gemm_kernel<<<gH, 256, 0, stream>>>(ybuf, Wo_t, bo, x, x1, nullptr, TOK, HID, HID, 2);

    // 6) LN2
    ln_kernel<<<TOK, 32, 0, stream>>>(x1, ln2g, ln2b, h2);

    // 7) MLP up + GELU
    gemm_kernel<<<dim3(FF / 64, TOK / 128), 256, 0, stream>>>(
        h2, W1_t, b1, nullptr, nullptr, gbuf, TOK, FF, HID, 1);

    // 8) MLP down + residual -> d_out (fp32)
    gemm_kernel<<<gH, 256, 0, stream>>>(gbuf, W2_t, b2, x1, (float*)d_out, nullptr, TOK, HID, FF, 2);
}
